// ParallelCoattention_10118942949447
// MI455X (gfx1250) — matrix operand order repaired and checked by reading, not compiled, not run on hardware
//
#include <hip/hip_runtime.h>
#include <cmath>

// ---------------------------------------------------------------------------
// ParallelCoattention forward for MI455X (gfx1250, wave32, WMMA)
//   out[b,n] = softmax_n( tanh( max_m ( (embT @ W_b @ emb)[b,n,m] ) ) )
// Fused: per (b, 16-row) tile -> P slab in LDS -> stream m via WMMA -> rowmax
// v2: dual independent accumulator chains per wave (shared A operand) to
//     pipeline the WMMA unit instead of serializing on one D->C chain.
// ---------------------------------------------------------------------------

typedef __attribute__((ext_vector_type(16))) __bf16 v16bf;
typedef __attribute__((ext_vector_type(8)))  __bf16 v8bf;
typedef __attribute__((ext_vector_type(8)))  float  v8f;

#define D_DIM 1024
#define B_DIM 16

static __device__ __forceinline__ v16bf cat16(v8bf lo, v8bf hi) {
  return __builtin_shufflevector(lo, hi, 0,1,2,3,4,5,6,7,8,9,10,11,12,13,14,15);
}

static __device__ __forceinline__ v8f wmma_bf16(v16bf a, v16bf b, v8f c) {
  return __builtin_amdgcn_wmma_f32_16x16x32_bf16(false, a, false, b,
                                                 (short)0, c, false, false);
}

// Tiled transpose + f32->bf16 convert: in [z][R][C] f32  ->  out [z][C][R] bf16
__global__ __launch_bounds__(256) void tconv_kernel(const float* __restrict__ in,
                                                    __bf16* __restrict__ out,
                                                    int R, int C) {
  __shared__ __bf16 tile[32][34];   // 17-bank stride: conflict-free transpose
  const int c0 = blockIdx.x * 32;
  const int r0 = blockIdx.y * 32;
  const size_t zoff = (size_t)blockIdx.z * (size_t)R * (size_t)C;
  const float* inb = in + zoff;
  __bf16* outb = out + zoff;
  const int tx = threadIdx.x;
  #pragma unroll
  for (int i = threadIdx.y; i < 32; i += 8)
    tile[i][tx] = (__bf16)inb[(size_t)(r0 + i) * C + (c0 + tx)];
  __syncthreads();
  #pragma unroll
  for (int i = threadIdx.y; i < 32; i += 8)
    outb[(size_t)(c0 + i) * R + (r0 + tx)] = tile[tx][i];
}

// Fused dual-GEMM + row-max.  1 workgroup = (batch b, rows n0..n0+15).
// 128 threads = 4 waves; wave w owns k-range [256w,256w+256) in phase 1 and
// every 4th m-tile-pair in phase 2.  Each wave runs TWO accumulator chains
// sharing one A operand (one A fetch feeds two v_wmma).
__global__ __launch_bounds__(128) void fused_gemm_max_kernel(
    const __bf16* __restrict__ EbfT,   // [B][N][D]  (= emb transposed, bf16)
    const __bf16* __restrict__ WbfT,   // [K][D]     (= W_b transposed, bf16)
    float* __restrict__ rowmaxg) {     // [B][N]  tanh(row max)
  __shared__ alignas(16) __bf16 P_lds[16][1032];   // 16 rows x 1024 (+8 pad)
  __shared__ float redbuf[16][64];

  const int b    = blockIdx.x >> 6;
  const int n0   = (blockIdx.x & 63) << 4;
  const int tid  = threadIdx.x;
  const int wv   = tid >> 5;          // wave id 0..3
  const int lane = tid & 31;
  const int hlf  = lane >> 4;         // half-wave selects K sub-chunk
  const int lr   = lane & 15;         // row (A) / column (B) index

  const __bf16* Eb   = EbfT + ((size_t)b << 20);          // b * 1024 * 1024
  const __bf16* Arow = Eb + (size_t)(n0 + lr) * D_DIM;    // embT row n0+lr
  const __bf16* a1   = Arow + hlf * 8;                    // K = d0 + 8h .. +7
  const __bf16* a2   = Arow + 16 + hlf * 8;               // K = d0+16+8h .. +7

  // ---- Phase 1: P[n][k] = sum_d embT[n][d] * W[d][k]  (bf16 into LDS) ----
  for (int t = 0; t < 8; ++t) {
    const int kbase = wv * 256 + t * 32;                  // k-tile pair
    const __bf16* B0 = WbfT + (size_t)(kbase + lr) * D_DIM + hlf * 8;
    const __bf16* B1 = B0 + (size_t)16 * D_DIM;
    v8f acc0{}, acc1{};
    for (int d0 = 0; d0 < D_DIM; d0 += 32) {
      v16bf A   = cat16(*(const v8bf*)(a1 + d0), *(const v8bf*)(a2 + d0));
      v16bf Bv0 = cat16(*(const v8bf*)(B0 + d0), *(const v8bf*)(B0 + d0 + 16));
      v16bf Bv1 = cat16(*(const v8bf*)(B1 + d0), *(const v8bf*)(B1 + d0 + 16));
      acc0 = wmma_bf16(A, Bv0, acc0);        // independent chains: pipeline
      acc1 = wmma_bf16(A, Bv1, acc1);
    }
    // C/D layout: VGPR i -> row i (lanes 0-15) / i+8 (lanes 16-31), col = lr
    #pragma unroll
    for (int i = 0; i < 8; ++i) {
      P_lds[hlf * 8 + i][kbase + lr]      = (__bf16)acc0[i];
      P_lds[hlf * 8 + i][kbase + 16 + lr] = (__bf16)acc1[i];
    }
  }
  __syncthreads();

  // ---- Phase 2: S[n][m] = sum_k P[n][k] * emb[b][k][m]; running row max ----
  const __bf16* p1 = &P_lds[lr][hlf * 8];
  const __bf16* p2 = &P_lds[lr][16 + hlf * 8];
  v8f runmax;
  #pragma unroll
  for (int i = 0; i < 8; ++i) runmax[i] = -3.0e38f;

  for (int t = 0; t < 8; ++t) {
    const int mbase = (t * 4 + wv) * 32;                  // m-tile pair
    const __bf16* B0 = Eb + (size_t)(mbase + lr) * D_DIM + hlf * 8;
    const __bf16* B1 = B0 + (size_t)16 * D_DIM;
    v8f acc0{}, acc1{};
    for (int k0 = 0; k0 < D_DIM; k0 += 32) {
      __builtin_prefetch(B0 + k0 + 128, 0, 0);   // global_prefetch_b8
      v16bf A   = cat16(*(const v8bf*)(p1 + k0), *(const v8bf*)(p2 + k0));
      v16bf Bv0 = cat16(*(const v8bf*)(B0 + k0), *(const v8bf*)(B0 + k0 + 16));
      v16bf Bv1 = cat16(*(const v8bf*)(B1 + k0), *(const v8bf*)(B1 + k0 + 16));
      acc0 = wmma_bf16(A, Bv0, acc0);
      acc1 = wmma_bf16(A, Bv1, acc1);
    }
    #pragma unroll
    for (int i = 0; i < 8; ++i) {
      runmax[i] = fmaxf(runmax[i], acc0[i]);   // max folds both tile columns
      runmax[i] = fmaxf(runmax[i], acc1[i]);
    }
  }

  // cross-wave max reduction (every slot written exactly once -> no init)
  #pragma unroll
  for (int i = 0; i < 8; ++i)
    redbuf[hlf * 8 + i][wv * 16 + lr] = runmax[i];
  __syncthreads();
  if (tid < 16) {
    float m = -3.0e38f;
    #pragma unroll 8
    for (int j = 0; j < 64; ++j) m = fmaxf(m, redbuf[tid][j]);
    rowmaxg[(size_t)b * D_DIM + n0 + tid] = tanhf(m);
  }
}

// Per-batch softmax over N=1024 values.
__global__ __launch_bounds__(256) void softmax_kernel(const float* __restrict__ rowmax,
                                                      float* __restrict__ out) {
  __shared__ float red[256];
  const int b = blockIdx.x;
  const int tid = threadIdx.x;
  const float* x = rowmax + (size_t)b * D_DIM;
  float v[4];
  float lmax = -3.0e38f;
  #pragma unroll
  for (int i = 0; i < 4; ++i) { v[i] = x[tid + i * 256]; lmax = fmaxf(lmax, v[i]); }
  red[tid] = lmax;
  __syncthreads();
  for (int s = 128; s > 0; s >>= 1) {
    if (tid < s) red[tid] = fmaxf(red[tid], red[tid + s]);
    __syncthreads();
  }
  const float m = red[0];
  __syncthreads();
  float e[4];
  float lsum = 0.f;
  #pragma unroll
  for (int i = 0; i < 4; ++i) { e[i] = expf(v[i] - m); lsum += e[i]; }
  red[tid] = lsum;
  __syncthreads();
  for (int s = 128; s > 0; s >>= 1) {
    if (tid < s) red[tid] += red[tid + s];
    __syncthreads();
  }
  const float inv = 1.f / red[0];
  #pragma unroll
  for (int i = 0; i < 4; ++i) out[(size_t)b * D_DIM + tid + i * 256] = e[i] * inv;
}

extern "C" void kernel_launch(void* const* d_in, const int* in_sizes, int n_in,
                              void* d_out, int out_size, void* d_ws, size_t ws_size,
                              hipStream_t stream) {
  const float* emb = (const float*)d_in[0];   // [B, D, N] f32
  // d_in[1] = heatm: unused by the forward pass
  const float* Wb  = (const float*)d_in[2];   // [D, D] f32

  const size_t EBT_BYTES = (size_t)B_DIM * D_DIM * D_DIM * 2u;  // 32 MB bf16
  const size_t WBT_BYTES = (size_t)D_DIM * D_DIM * 2u;          //  2 MB bf16
  __bf16* EbfT  = (__bf16*)d_ws;
  __bf16* WbfT  = (__bf16*)((char*)d_ws + EBT_BYTES);
  float* rowmax = (float*)((char*)d_ws + EBT_BYTES + WBT_BYTES);

  dim3 blkT(32, 8, 1);
  // emb [b][d][n] -> EbfT [b][n][d]
  tconv_kernel<<<dim3(D_DIM / 32, D_DIM / 32, B_DIM), blkT, 0, stream>>>(
      emb, EbfT, D_DIM, D_DIM);
  // W_b [d][k] -> WbfT [k][d]
  tconv_kernel<<<dim3(D_DIM / 32, D_DIM / 32, 1), blkT, 0, stream>>>(
      Wb, WbfT, D_DIM, D_DIM);

  fused_gemm_max_kernel<<<dim3(B_DIM * (D_DIM / 16)), dim3(128), 0, stream>>>(
      EbfT, WbfT, rowmax);

  softmax_kernel<<<dim3(B_DIM), dim3(256), 0, stream>>>(rowmax, (float*)d_out);
}
